// LSTMCell_82669530514116
// MI455X (gfx1250) — matrix operand order repaired and checked by reading, not compiled, not run on hardware
//
#include <hip/hip_runtime.h>

// ---------------- problem constants ----------------
#define B_DIM  16384
#define IN_DIM 512
#define H_DIM  512
#define K_DIM  1024          // IN + H
#define NGATE  4

// GEMM tiling
#define BM 128               // rows per block
#define BN 32                // gate-cols (per gate) per block
#define BK 32                // k per step
#define LDA 40               // LDS row stride (bf16 elems), padded
#define LDB 40

typedef __attribute__((ext_vector_type(4)))  float          f32x4;
typedef __attribute__((ext_vector_type(8)))  float          f32x8;
typedef __attribute__((ext_vector_type(4)))  int            i32x4;
typedef __attribute__((ext_vector_type(8)))  unsigned short u16x8;
typedef __attribute__((ext_vector_type(16))) unsigned short u16x16;
typedef __attribute__((ext_vector_type(16))) __bf16         bf16x16;

// ---- CDNA5 async global->LDS path (ASYNCcnt), with safe fallback ----------
#if __has_builtin(__builtin_amdgcn_global_load_async_to_lds_b128)
#define USE_ASYNC_LDS 1
#endif

typedef __attribute__((address_space(1))) i32x4 gl_i32x4;   // global int4
typedef __attribute__((address_space(3))) i32x4 lds_i32x4;  // LDS int4

static __device__ __forceinline__ void cp16_g2l(const unsigned short* g,
                                                unsigned short* l) {
#if defined(USE_ASYNC_LDS)
  __builtin_amdgcn_global_load_async_to_lds_b128(
      (gl_i32x4*)g, (lds_i32x4*)l, 0, 0);
#else
  *(u16x8*)l = *(const u16x8*)g;     // sync fallback: load + ds_store
#endif
}

static __device__ __forceinline__ void wait_async_zero() {
#if defined(USE_ASYNC_LDS)
#if __has_builtin(__builtin_amdgcn_s_wait_asynccnt)
  __builtin_amdgcn_s_wait_asynccnt(0);
#else
  asm volatile("s_wait_asynccnt 0x0" ::: "memory");
#endif
#endif
}

static __device__ __forceinline__ unsigned short f2bf(float f) {
  unsigned int x = __builtin_bit_cast(unsigned int, f);
  if ((x & 0x7fffffffu) > 0x7f800000u)           // NaN -> quiet NaN
    return (unsigned short)((x >> 16) | 0x0040u);
  return (unsigned short)((x + 0x7fffu + ((x >> 16) & 1u)) >> 16);  // RNE
}

// ---------------------------------------------------------------------------
// Pack kernel 1: Abf[b][k] = bf16( k < 512 ? x[b][k] : h[b][k-512] )
// ---------------------------------------------------------------------------
__global__ __launch_bounds__(256) void lstm_pack_a(const float* __restrict__ x,
                                                   const float* __restrict__ h,
                                                   unsigned short* __restrict__ Abf) {
  int idx = blockIdx.x * 256 + threadIdx.x;        // [0, B*K/8)
  int k8  = idx & 127;                             // k = k8*8
  int b   = idx >> 7;
  const float* src = (k8 < 64) ? (x + (size_t)b * IN_DIM + k8 * 8)
                               : (h + (size_t)b * H_DIM + (k8 - 64) * 8);
  f32x4 lo = *(const f32x4*)src;
  f32x4 hi = *(const f32x4*)(src + 4);
  u16x8 o;
  o[0] = f2bf(lo[0]); o[1] = f2bf(lo[1]); o[2] = f2bf(lo[2]); o[3] = f2bf(lo[3]);
  o[4] = f2bf(hi[0]); o[5] = f2bf(hi[1]); o[6] = f2bf(hi[2]); o[7] = f2bf(hi[3]);
  *(u16x8*)(Abf + (size_t)b * K_DIM + k8 * 8) = o;
}

// ---------------------------------------------------------------------------
// Pack kernel 2: Wt[g][n][k] = bf16( W_g[k][n] )  (transposed weights so each
// WMMA B-fragment is contiguous-K per output column)
// ---------------------------------------------------------------------------
__global__ __launch_bounds__(256) void lstm_pack_w(const float* __restrict__ Wi,
                                                   const float* __restrict__ Wf,
                                                   const float* __restrict__ Wo,
                                                   const float* __restrict__ Wc,
                                                   unsigned short* __restrict__ Wt) {
  int idx = blockIdx.x * 256 + threadIdx.x;        // [0, 4*1024*512)
  int n   = idx & 511;
  int kk  = (idx >> 9) & 1023;
  int g   = idx >> 19;
  const float* Wg = (g == 0) ? Wi : (g == 1) ? Wf : (g == 2) ? Wo : Wc;
  float w = Wg[(size_t)kk * H_DIM + n];
  Wt[((size_t)(g * H_DIM + n)) * K_DIM + kk] = f2bf(w);
}

// ---------------------------------------------------------------------------
// Fused 4-gate GEMM + LSTM epilogue, double-buffered async LDS staging.
// Block: 256 threads = 8 waves; tile BM=128 rows x BN=32 cols x 4 gates.
// ---------------------------------------------------------------------------
__global__ __launch_bounds__(256) void lstm_gemm_wmma(
    const unsigned short* __restrict__ Abf,
    const unsigned short* __restrict__ Wt,
    const float* __restrict__ bi, const float* __restrict__ bfv,
    const float* __restrict__ bo, const float* __restrict__ bc,
    const float* __restrict__ c_cur, float* __restrict__ out) {

  __shared__ __attribute__((aligned(16))) unsigned short As[2][BM * LDA];
  __shared__ __attribute__((aligned(16))) unsigned short Bs[2][NGATE * BN * LDB];

  const int t    = threadIdx.x;
  const int lane = t & 31;
  const int wave = t >> 5;
  const int wm   = wave & 3;        // 0..3 -> 32 rows each
  const int wn   = wave >> 2;       // 0..1 -> 16 cols each
  const int m0   = blockIdx.x * BM;
  const int n0   = blockIdx.y * BN;

  const int hh = lane >> 4;         // half-wave 0/1
  const int lr = lane & 15;

  // staging indices (per thread: 2x16B for A, 2x16B for B per k-step)
  const int rA = t >> 1;               // 0..127
  const int hA = (t & 1) * 16;         // 0 or 16
  const int gB = t >> 6;               // gate 0..3
  const int nB = (t & 63) >> 1;        // 0..31
  const int hB = ((t & 63) & 1) * 16;  // 0 or 16

  const unsigned short* gA = Abf + (size_t)(m0 + rA) * K_DIM + hA;
  const unsigned short* gW = Wt + ((size_t)(gB * H_DIM + n0 + nB)) * K_DIM + hB;

  f32x8 acc[NGATE][2] = {};

  // prologue: stage tile 0 into buffer 0
  {
    unsigned short* lA = &As[0][rA * LDA + hA];
    unsigned short* lB = &Bs[0][gB * (BN * LDB) + nB * LDB + hB];
    cp16_g2l(gA,     lA);
    cp16_g2l(gA + 8, lA + 8);
    cp16_g2l(gW,     lB);
    cp16_g2l(gW + 8, lB + 8);
  }

  int buf = 0;
  for (int k0 = 0; k0 < K_DIM; k0 += BK) {
    wait_async_zero();        // my async copies into As/Bs[buf] are done
    __syncthreads();          // everyone's are done; prior compute also done

    // stage next tile into the other buffer (overlaps with this tile's WMMAs)
    if (k0 + BK < K_DIM) {
      int nb = buf ^ 1;
      const unsigned short* pA = gA + (k0 + BK);
      const unsigned short* pW = gW + (k0 + BK);
      unsigned short* lA = &As[nb][rA * LDA + hA];
      unsigned short* lB = &Bs[nb][gB * (BN * LDB) + nB * LDB + hB];
      cp16_g2l(pA,     lA);
      cp16_g2l(pA + 8, lA + 8);
      cp16_g2l(pW,     lB);
      cp16_g2l(pW + 8, lB + 8);
    }

    // ---- load ALL fragments first (batch the ds_loads, amortize DScnt) ----
    // A fragment (16x32 bf16): lanes 0-15 row M=lr K{0-15},
    // lanes 16-31 same rows K{16-31}  (contiguous, matches B)
    bf16x16 afrag[2];
    bf16x16 bfrag[NGATE];
#pragma unroll
    for (int mt = 0; mt < 2; ++mt) {
      const unsigned short* p = &As[buf][(wm * 32 + mt * 16 + lr) * LDA + hh * 16];
      u16x8 alo = *(const u16x8*)p;
      u16x8 ahi = *(const u16x8*)(p + 8);
      u16x16 a  = __builtin_shufflevector(alo, ahi, 0,1,2,3,4,5,6,7,8,9,10,11,12,13,14,15);
      afrag[mt] = __builtin_bit_cast(bf16x16, a);
    }
#pragma unroll
    for (int g = 0; g < NGATE; ++g) {
      // B fragment (32x16 bf16): lane holds column N=lr, K=(hh*16)..+15 contiguous
      const unsigned short* p =
          &Bs[buf][g * (BN * LDB) + (wn * 16 + lr) * LDB + hh * 16];
      u16x8 blo = *(const u16x8*)p;
      u16x8 bhi = *(const u16x8*)(p + 8);
      u16x16 bb = __builtin_shufflevector(blo, bhi, 0,1,2,3,4,5,6,7,8,9,10,11,12,13,14,15);
      bfrag[g] = __builtin_bit_cast(bf16x16, bb);
    }

    // ---- 8 back-to-back WMMAs ----
#pragma unroll
    for (int g = 0; g < NGATE; ++g) {
      acc[g][0] = __builtin_amdgcn_wmma_f32_16x16x32_bf16(
          false, afrag[0], false, bfrag[g], (short)0, acc[g][0], false, false);
      acc[g][1] = __builtin_amdgcn_wmma_f32_16x16x32_bf16(
          false, afrag[1], false, bfrag[g], (short)0, acc[g][1], false, false);
    }
    buf ^= 1;
  }

  // ---- fused LSTM epilogue ----
  // C/D layout: VGPR r, lane l -> M = r + 8*(l>=16), N = l&15  (ISA 7.12.2)
  const int ncol = n0 + wn * 16 + lr;
  const float bi_v = bi[ncol], bf_v = bfv[ncol], bo_v = bo[ncol], bc_v = bc[ncol];
#pragma unroll
  for (int mt = 0; mt < 2; ++mt) {
#pragma unroll
    for (int r = 0; r < 8; ++r) {
      int mrow = m0 + wm * 32 + mt * 16 + r + hh * 8;
      float gi = acc[0][mt][r] + bi_v;
      float gf = acc[1][mt][r] + bf_v;
      float go = acc[2][mt][r] + bo_v;
      float gc = acc[3][mt][r] + bc_v;
      float ig = 1.0f / (1.0f + expf(-gi));
      float fg = 1.0f / (1.0f + expf(-gf));
      float og = 1.0f / (1.0f + expf(-go));
      float cg = tanhf(gc);
      size_t off = (size_t)mrow * H_DIM + ncol;
      float cn = fg * c_cur[off] + ig * cg;
      float hn = og * tanhf(cn);
      out[off] = hn;                                  // h_next
      out[(size_t)B_DIM * H_DIM + off] = cn;          // c_next
    }
  }
}

// ---------------------------------------------------------------------------
extern "C" void kernel_launch(void* const* d_in, const int* in_sizes, int n_in,
                              void* d_out, int out_size, void* d_ws, size_t ws_size,
                              hipStream_t stream) {
  const float* x  = (const float*)d_in[0];
  const float* h  = (const float*)d_in[1];
  const float* c  = (const float*)d_in[2];
  const float* Wi = (const float*)d_in[3];
  const float* bi = (const float*)d_in[4];
  const float* Wf = (const float*)d_in[5];
  const float* bf = (const float*)d_in[6];
  const float* Wo = (const float*)d_in[7];
  const float* bo = (const float*)d_in[8];
  const float* Wc = (const float*)d_in[9];
  const float* bc = (const float*)d_in[10];
  float* out = (float*)d_out;

  // workspace: 32 MB packed activations + 4 MB packed (transposed) weights
  unsigned short* Abf = (unsigned short*)d_ws;                 // [B][1024] bf16
  unsigned short* Wt  = Abf + (size_t)B_DIM * K_DIM;           // [4*512][1024] bf16

  lstm_pack_a<<<(B_DIM * K_DIM / 8) / 256, 256, 0, stream>>>(x, h, Abf);
  lstm_pack_w<<<(NGATE * K_DIM * H_DIM) / 256, 256, 0, stream>>>(Wi, Wf, Wo, Wc, Wt);

  dim3 grid(B_DIM / BM, H_DIM / BN);
  lstm_gemm_wmma<<<grid, 256, 0, stream>>>(Abf, Wt, bi, bf, bo, bc, c, out);
}
